// DeformBottleneck_28647431864870
// MI455X (gfx1250) — matrix operand order repaired and checked by reading, not compiled, not run on hardware
//
#include <hip/hip_runtime.h>
#include <hip/hip_bf16.h>

// ---------------------------------------------------------------------------
// Problem constants
// ---------------------------------------------------------------------------
#define BB    8
#define C1    256
#define C2    256
#define CM    128
#define HH    80
#define WW    80
#define HW    (HH * WW)        // 6400
#define NK    9                // 3x3 taps
#define KDIM  (CM * NK)        // 1152  (deform GEMM K)
#define PIXT  16               // pixels per tile
#define NTILE (HW / PIXT)      // 400

// padded LDS row strides (halfwords): multiple of 8 for 16B alignment,
// +8 padding de-conflicts banks (stride/4 dwords = 4 mod 64)
#define ST1   (C1 + 8)         // 264
#define ST3   (KDIM + 8)       // 1160
#define ST4   (CM + 8)         // 136

// ---------------------------------------------------------------------------
// WMMA vector types (CDNA5 gfx1250, wave32)
// ---------------------------------------------------------------------------
typedef __attribute__((ext_vector_type(16))) __bf16 v16bf;
typedef __attribute__((ext_vector_type(8)))  __bf16 v8bf;
typedef __attribute__((ext_vector_type(8)))  float  v8f;

union V16BF {
    v16bf v;
    v8bf  h[2];
    unsigned short u[16];
};

__device__ __forceinline__ unsigned short f32_to_bf16(float f) {
    unsigned u = __builtin_bit_cast(unsigned, f);
    unsigned r = u + 0x7FFFu + ((u >> 16) & 1u);   // round-to-nearest-even
    return (unsigned short)(r >> 16);
}

__device__ __forceinline__ float silu(float x) {
    return x * (1.0f / (1.0f + __expf(-x)));
}

// A-fragment (16x32 bf16) from row-major bf16 weights in global, stride ld.
// Per ISA 7.12.2: M = lane&15 for both halves; lane<16 -> K base 0, lane>=16 -> +8.
// Fragment elements: e=0..7 -> K=koff+e, e=8..15 -> K=16+koff+(e-8)
// => two contiguous 16-byte runs -> two b128 global loads.
__device__ __forceinline__ v16bf load_a_g(const unsigned short* __restrict__ W,
                                          int row0, int ld, int k0, int lane) {
    V16BF a;
    const int m    = lane & 15;
    const int koff = (lane < 16) ? 0 : 8;
    const unsigned short* wr = W + (size_t)(row0 + m) * ld + k0 + koff;
    a.h[0] = *(const v8bf*)(wr);
    a.h[1] = *(const v8bf*)(wr + 16);
    return a.v;
}

// B-fragment (32x16 bf16) from LDS stored n-major: S[n*stride + k].
// Lane's fragment: N = lane&15, koff = (lane<16?0:8); e=0..7 -> K=k0+koff+e, e=8..15 -> K=k0+16+koff+(e-8)
// => two contiguous 16-byte runs -> two aligned 16-byte LDS loads.
__device__ __forceinline__ v16bf load_b_lds(const unsigned short* S, int stride,
                                            int k0, int lane) {
    V16BF b;
    const int n    = lane & 15;
    const int koff = (lane < 16) ? 0 : 8;
    const unsigned short* p = S + n * stride + k0 + koff;
    b.h[0] = *(const v8bf*)(p);
    b.h[1] = *(const v8bf*)(p + 16);
    return b.v;
}

// ---------------------------------------------------------------------------
// Kernel 0: f32 -> bf16 weight conversion
// ---------------------------------------------------------------------------
__global__ void k_cvt_bf16(const float* __restrict__ src, unsigned short* __restrict__ dst, int n) {
    int i = blockIdx.x * 256 + threadIdx.x;
    if (i < n) dst[i] = f32_to_bf16(src[i]);
}

// ---------------------------------------------------------------------------
// Kernel 1: y1 = SiLU(BN1(conv1x1(x, cv1_w)))     GEMM M=128 K=256 N=HW
// grid (NTILE, B), block 256 (8 waves; wave w -> output channels 16w..16w+15)
// ---------------------------------------------------------------------------
__global__ void __launch_bounds__(256)
k_conv1_bn_silu(const float* __restrict__ x, const unsigned short* __restrict__ w1,
                const float* __restrict__ g, const float* __restrict__ bt,
                const float* __restrict__ mn, const float* __restrict__ vr,
                float* __restrict__ y1) {
    __shared__ __align__(32) unsigned short sX[PIXT * ST1];   // n-major, ~8.3 KB
    const int tile = blockIdx.x, b = blockIdx.y;
    const int p0 = tile * PIXT, tid = threadIdx.x;

    // cooperative load/convert of X tile: channel k = tid (C1 == 256 == blockDim)
    {
        const float* xr = x + ((size_t)b * C1 + tid) * HW + p0;
#pragma unroll
        for (int j = 0; j < PIXT; ++j) sX[j * ST1 + tid] = f32_to_bf16(xr[j]);
    }
    __syncthreads();

    const int wave = tid >> 5, lane = tid & 31;
    const int oc0 = wave * 16;
    v8f acc = {};
#pragma unroll
    for (int k0 = 0; k0 < C1; k0 += 32) {
        v16bf a  = load_a_g(w1, oc0, C1, k0, lane);
        v16bf bm = load_b_lds(sX, ST1, k0, lane);
        acc = __builtin_amdgcn_wmma_f32_16x16x32_bf16(false, a, false, bm,
                                                      (short)0, acc, false, false);
    }
    const int n = lane & 15, mrow = (lane >> 4) << 3;
#pragma unroll
    for (int r = 0; r < 8; ++r) {
        int oc = oc0 + mrow + r;
        float inv = g[oc] * rsqrtf(vr[oc] + 1e-5f);
        float val = acc[r] * inv + (bt[oc] - mn[oc] * inv);
        y1[((size_t)b * CM + oc) * HW + p0 + n] = silu(val);
    }
}

// ---------------------------------------------------------------------------
// Kernel 2: offset/mask prediction = conv3x3(y1, off_w) + off_b
// channels 0..17 raw offsets, 18..26 sigmoid mask. Direct fp32 (small op).
// ---------------------------------------------------------------------------
__global__ void __launch_bounds__(256)
k_offset_conv(const float* __restrict__ y1, const float* __restrict__ ow,
              const float* __restrict__ ob, float* __restrict__ om) {
    int idx = blockIdx.x * 256 + threadIdx.x;
    if (idx >= BB * 27 * HW) return;
    int p = idx % HW;
    int t = idx / HW;
    int oc = t % 27;
    int b  = t / 27;
    int h = p / WW, w = p % WW;

    float sum = ob[oc];
    const float* wr = ow + (size_t)oc * CM * NK;
    const float* yb = y1 + (size_t)b * CM * HW;
    for (int c = 0; c < CM; ++c) {
        const float* yc = yb + (size_t)c * HW;
        const float* wc = wr + c * NK;
#pragma unroll
        for (int ky = 0; ky < 3; ++ky) {
            int yy = h + ky - 1;
            if (yy < 0 || yy >= HH) continue;
#pragma unroll
            for (int kx = 0; kx < 3; ++kx) {
                int xx = w + kx - 1;
                if (xx < 0 || xx >= WW) continue;
                sum = fmaf(yc[yy * WW + xx], wc[ky * 3 + kx], sum);
            }
        }
    }
    if (oc >= 18) sum = 1.0f / (1.0f + __expf(-sum));
    om[((size_t)b * 27 + oc) * HW + p] = sum;
}

// ---------------------------------------------------------------------------
// Kernel 3: deformable conv = WMMA GEMM over bilinearly gathered im2col tile
//           + dcn bias + BN2 + SiLU.   M=128, K=1152, N=HW
// grid (NTILE, B), block 256.
// ---------------------------------------------------------------------------
__global__ void __launch_bounds__(256)
k_deform_gemm(const float* __restrict__ y1, const float* __restrict__ om,
              const unsigned short* __restrict__ wd, const float* __restrict__ dcnb,
              const float* __restrict__ g, const float* __restrict__ bt,
              const float* __restrict__ mn, const float* __restrict__ vr,
              float* __restrict__ y2) {
    __shared__ __align__(32) unsigned short sS[PIXT * ST3];   // n-major, ~36.3 KB
    __shared__ int   t_y0[PIXT * NK], t_x0[PIXT * NK];
    __shared__ float t_w[PIXT * NK * 4];

    const int tile = blockIdx.x, b = blockIdx.y;
    const int p0 = tile * PIXT, tid = threadIdx.x;

    // phase 1: per (pixel, tap) sampling geometry (mask folded into weights)
    if (tid < PIXT * NK) {
        int pix = tid / NK, tap = tid % NK;
        int p = p0 + pix, h = p / WW, w = p % WW;
        int ky = tap / 3, kx = tap % 3;
        const float* omb = om + (size_t)b * 27 * HW;
        float oy  = omb[(tap * 2 + 0) * HW + p];
        float ox  = omb[(tap * 2 + 1) * HW + p];
        float msk = omb[(18 + tap) * HW + p];
        float py = (float)(h - 1 + ky) + oy;
        float px = (float)(w - 1 + kx) + ox;
        float fy = floorf(py), fx = floorf(px);
        int y0 = (int)fy, x0 = (int)fx;
        float wy1 = py - fy, wx1 = px - fx;
        float wy0 = 1.0f - wy1, wx0 = 1.0f - wx1;
        float vy0 = (y0 >= 0 && y0 < HH) ? 1.0f : 0.0f;
        float vy1 = (y0 + 1 >= 0 && y0 + 1 < HH) ? 1.0f : 0.0f;
        float vx0 = (x0 >= 0 && x0 < WW) ? 1.0f : 0.0f;
        float vx1 = (x0 + 1 >= 0 && x0 + 1 < WW) ? 1.0f : 0.0f;
        t_y0[tid] = y0;
        t_x0[tid] = x0;
        t_w[tid * 4 + 0] = wy0 * wx0 * msk * vy0 * vx0;
        t_w[tid * 4 + 1] = wy0 * wx1 * msk * vy0 * vx1;
        t_w[tid * 4 + 2] = wy1 * wx0 * msk * vy1 * vx0;
        t_w[tid * 4 + 3] = wy1 * wx1 * msk * vy1 * vx1;
    }
    __syncthreads();

    // phase 2: gather + bilinear blend -> LDS im2col tile (bf16, n-major)
    for (int idx = tid; idx < CM * PIXT * NK; idx += 256) {
        int c  = idx / (PIXT * NK);
        int pt = idx % (PIXT * NK);
        int pix = pt / NK, tap = pt % NK;
        int y0 = t_y0[pt], x0 = t_x0[pt];
        int yc0 = min(max(y0, 0), HH - 1),     yc1 = min(max(y0 + 1, 0), HH - 1);
        int xc0 = min(max(x0, 0), WW - 1),     xc1 = min(max(x0 + 1, 0), WW - 1);
        const float* yp = y1 + ((size_t)b * CM + c) * HW;
        float gv = t_w[pt * 4 + 0] * yp[yc0 * WW + xc0]
                 + t_w[pt * 4 + 1] * yp[yc0 * WW + xc1]
                 + t_w[pt * 4 + 2] * yp[yc1 * WW + xc0]
                 + t_w[pt * 4 + 3] * yp[yc1 * WW + xc1];
        int k = c * NK + tap;
        sS[pix * ST3 + k] = f32_to_bf16(gv);
    }
    __syncthreads();

    // phase 3: WMMA GEMM, K = 1152 in 36 steps of 32
    const int wave = tid >> 5, lane = tid & 31;
    const int oc0 = wave * 16;
    v8f acc = {};
    for (int k0 = 0; k0 < KDIM; k0 += 32) {
        if (k0 + 32 < KDIM)  // stream next weight stripe (global_prefetch_b8)
            __builtin_prefetch(wd + (size_t)(oc0 + (lane & 15)) * KDIM + k0 + 32, 0, 3);
        v16bf a  = load_a_g(wd, oc0, KDIM, k0, lane);
        v16bf bm = load_b_lds(sS, ST3, k0, lane);
        acc = __builtin_amdgcn_wmma_f32_16x16x32_bf16(false, a, false, bm,
                                                      (short)0, acc, false, false);
    }
    const int n = lane & 15, mrow = (lane >> 4) << 3;
#pragma unroll
    for (int r = 0; r < 8; ++r) {
        int oc = oc0 + mrow + r;
        float inv = g[oc] * rsqrtf(vr[oc] + 1e-5f);
        float val = (acc[r] + dcnb[oc]) * inv + (bt[oc] - mn[oc] * inv);
        y2[((size_t)b * CM + oc) * HW + p0 + n] = silu(val);
    }
}

// ---------------------------------------------------------------------------
// Kernel 4: out = x + SiLU(BN3(conv1x1(y2, cv3_w)))   GEMM M=256 K=128 N=HW
// grid (NTILE, B), block 256; each wave handles two 16-channel stripes.
// ---------------------------------------------------------------------------
__global__ void __launch_bounds__(256)
k_conv3_bn_silu_res(const float* __restrict__ y2, const unsigned short* __restrict__ w3,
                    const float* __restrict__ g, const float* __restrict__ bt,
                    const float* __restrict__ mn, const float* __restrict__ vr,
                    const float* __restrict__ x, float* __restrict__ out) {
    __shared__ __align__(32) unsigned short sY[PIXT * ST4];   // n-major, ~4.3 KB
    const int tile = blockIdx.x, b = blockIdx.y;
    const int p0 = tile * PIXT, tid = threadIdx.x;

    if (tid < CM) {
        const float* yr = y2 + ((size_t)b * CM + tid) * HW + p0;
#pragma unroll
        for (int j = 0; j < PIXT; ++j) sY[j * ST4 + tid] = f32_to_bf16(yr[j]);
    }
    __syncthreads();

    const int wave = tid >> 5, lane = tid & 31;
    const int n = lane & 15, mrow = (lane >> 4) << 3;
#pragma unroll
    for (int half = 0; half < 2; ++half) {
        int oc0 = wave * 32 + half * 16;
        v8f acc = {};
#pragma unroll
        for (int k0 = 0; k0 < CM; k0 += 32) {
            v16bf a  = load_a_g(w3, oc0, CM, k0, lane);
            v16bf bm = load_b_lds(sY, ST4, k0, lane);
            acc = __builtin_amdgcn_wmma_f32_16x16x32_bf16(false, a, false, bm,
                                                          (short)0, acc, false, false);
        }
#pragma unroll
        for (int r = 0; r < 8; ++r) {
            int oc = oc0 + mrow + r;
            float inv = g[oc] * rsqrtf(vr[oc] + 1e-5f);
            float val = acc[r] * inv + (bt[oc] - mn[oc] * inv);
            size_t o = ((size_t)b * C2 + oc) * HW + p0 + n;
            out[o] = x[o] + silu(val);
        }
    }
}

// ---------------------------------------------------------------------------
// Workspace layout (bytes)
// ---------------------------------------------------------------------------
#define WS_W1   0ull                        // 128*256  bf16 =   64 KB
#define WS_W3   65536ull                    // 256*128  bf16 =   64 KB
#define WS_WD   131072ull                   // 128*1152 bf16 =  288 KB
#define WS_Y1   524288ull                   // 8*128*6400 f32 = 26.2 MB
#define WS_OM   (WS_Y1 + 26214400ull)       // 8*27*6400  f32 =  5.5 MB
#define WS_Y2   (WS_OM + 5529600ull)        // 8*128*6400 f32 = 26.2 MB

extern "C" void kernel_launch(void* const* d_in, const int* in_sizes, int n_in,
                              void* d_out, int out_size, void* d_ws, size_t ws_size,
                              hipStream_t stream) {
    const float* x     = (const float*)d_in[0];
    const float* cv1_w = (const float*)d_in[1];
    const float* bn1_g = (const float*)d_in[2];
    const float* bn1_b = (const float*)d_in[3];
    const float* bn1_m = (const float*)d_in[4];
    const float* bn1_v = (const float*)d_in[5];
    const float* off_w = (const float*)d_in[6];
    const float* off_b = (const float*)d_in[7];
    const float* dcn_w = (const float*)d_in[8];
    const float* dcn_b = (const float*)d_in[9];
    const float* bn2_g = (const float*)d_in[10];
    const float* bn2_b = (const float*)d_in[11];
    const float* bn2_m = (const float*)d_in[12];
    const float* bn2_v = (const float*)d_in[13];
    const float* cv3_w = (const float*)d_in[14];
    const float* bn3_g = (const float*)d_in[15];
    const float* bn3_b = (const float*)d_in[16];
    const float* bn3_m = (const float*)d_in[17];
    const float* bn3_v = (const float*)d_in[18];

    char* ws = (char*)d_ws;
    unsigned short* w1b = (unsigned short*)(ws + WS_W1);
    unsigned short* w3b = (unsigned short*)(ws + WS_W3);
    unsigned short* wdb = (unsigned short*)(ws + WS_WD);
    float* y1 = (float*)(ws + WS_Y1);
    float* om = (float*)(ws + WS_OM);
    float* y2 = (float*)(ws + WS_Y2);

    // weight conversion to bf16
    k_cvt_bf16<<<128, 256, 0, stream>>>(cv1_w, w1b, CM * C1);          // 32768
    k_cvt_bf16<<<128, 256, 0, stream>>>(cv3_w, w3b, C2 * CM);          // 32768
    k_cvt_bf16<<<576, 256, 0, stream>>>(dcn_w, wdb, CM * KDIM);        // 147456

    dim3 grid(NTILE, BB);
    k_conv1_bn_silu<<<grid, 256, 0, stream>>>(x, w1b, bn1_g, bn1_b, bn1_m, bn1_v, y1);
    k_offset_conv<<<(BB * 27 * HW) / 256, 256, 0, stream>>>(y1, off_w, off_b, om);
    k_deform_gemm<<<grid, 256, 0, stream>>>(y1, om, wdb, dcn_b,
                                            bn2_g, bn2_b, bn2_m, bn2_v, y2);
    k_conv3_bn_silu_res<<<grid, 256, 0, stream>>>(y2, w3b, bn3_g, bn3_b, bn3_m, bn3_v,
                                                  x, (float*)d_out);
}
